// SSLPretrainModel_53944789238404
// MI455X (gfx1250) — matrix operand order repaired and checked by reading, not compiled, not run on hardware
//
#include <hip/hip_runtime.h>
#include <hip/hip_bf16.h>

// ---------------------------------------------------------------------------
// D-MPNN (Chemprop SSL pretrain) for MI455X / gfx1250, wave32 + WMMA f16.
// ---------------------------------------------------------------------------

typedef __attribute__((ext_vector_type(16))) _Float16 v16h;
typedef __attribute__((ext_vector_type(8)))  _Float16 h8;
typedef __attribute__((ext_vector_type(4)))  _Float16 h4;
typedef __attribute__((ext_vector_type(8)))  float    v8f;
typedef __attribute__((ext_vector_type(4)))  float    f4;

#define N_ATOMS 200000
#define N_BONDS 400000
#define N_MOLS  8192
#define HIDDEN  300
#define ATOM_FDIM 133
#define BOND_FDIM 147
#define HG4 (HIDDEN / 4)   // 75 column-groups of 4

// ---------------------------------------------------------------------------
// Utility kernels
// ---------------------------------------------------------------------------
__global__ __launch_bounds__(256) void zero_f32_kernel(float* p, long long n) {
    long long i = (long long)blockIdx.x * blockDim.x + threadIdx.x;
    if (i < n) p[i] = 0.0f;
}
__global__ __launch_bounds__(256) void zero_f16_kernel(_Float16* p, long long n) {
    long long i = (long long)blockIdx.x * blockDim.x + threadIdx.x;
    if (i < n) p[i] = (_Float16)0.0f;
}

// f32 [rows, Kvalid] row-major -> f16 [rows, ld_dst] with zero pad cols.
__global__ __launch_bounds__(256) void conv_pad_kernel(const float* __restrict__ src,
                                                       _Float16* __restrict__ dst,
                                                       int rows, int Kvalid, int ld_dst) {
    long long i = (long long)blockIdx.x * blockDim.x + threadIdx.x;
    long long total = (long long)rows * ld_dst;
    if (i >= total) return;
    int r = (int)(i / ld_dst);
    int k = (int)(i % ld_dst);
    float v = (k < Kvalid) ? src[(long long)r * Kvalid + k] : 0.0f;
    dst[i] = (_Float16)v;
}

// bias f32 [Nvalid] -> padded f32 [Ntot] (zeros beyond Nvalid)
__global__ __launch_bounds__(256) void pad_bias_kernel(const float* __restrict__ src,
                                                       float* __restrict__ dst,
                                                       int Nvalid, int Ntot) {
    int i = blockIdx.x * blockDim.x + threadIdx.x;
    if (i < Ntot) dst[i] = (i < Nvalid) ? src[i] : 0.0f;
}

// Pack weight W [Nout, Kin] (row-major, f32) into per-lane WMMA-B layout:
// Bp[(((nt*KT)+t)*32 + lane)*16 + e] = W[n, k] with
//   n = nt*16 + (lane & 15),  k = t*32 + ((lane>>4)*16) + e   (zero padded)
__global__ __launch_bounds__(256) void pack_weights_kernel(const float* __restrict__ W,
                                                           int Nout, int Kin,
                                                           int KT, int n_tiles,
                                                           _Float16* __restrict__ Bp) {
    long long i = (long long)blockIdx.x * blockDim.x + threadIdx.x;
    long long total = (long long)n_tiles * KT * 32 * 16;
    if (i >= total) return;
    int e    = (int)(i & 15);
    int lane = (int)((i >> 4) & 31);
    int t    = (int)((i >> 9) % KT);
    int nt   = (int)(i / ((long long)KT * 512));
    int n = nt * 16 + (lane & 15);
    int k = t * 32 + ((lane >> 4) * 16) + e;
    _Float16 v = (_Float16)0.0f;
    if (n < Nout && k < Kin) v = (_Float16)W[(long long)n * Kin + k];
    Bp[i] = v;
}

// Segment scatter-add, 4 columns per thread:
//   dst[idx[r], 4g..4g+3] += (float)src[r, 4g..4g+3]
__global__ __launch_bounds__(256) void scatter_add_kernel(const _Float16* __restrict__ src,
                                                          int ld_src,
                                                          const int* __restrict__ idx,
                                                          float* __restrict__ dst,
                                                          int rows) {
    long long i = (long long)blockIdx.x * blockDim.x + threadIdx.x;
    long long total = (long long)rows * HG4;
    if (i >= total) return;
    int r = (int)(i / HG4);
    int g = (int)(i % HG4);
    h4 v = *(const h4*)(src + (long long)r * ld_src + 4 * g);
    float* d = dst + (long long)idx[r] * HIDDEN + 4 * g;
    unsafeAtomicAdd(d + 0, (float)v[0]);
    unsafeAtomicAdd(d + 1, (float)v[1]);
    unsafeAtomicAdd(d + 2, (float)v[2]);
    unsafeAtomicAdd(d + 3, (float)v[3]);
}

// message[b, 4g..4g+3] = atom_sum[b2a[b], ...] - bond_hidden[b^1, ...]
__global__ __launch_bounds__(256) void build_message_kernel(const float* __restrict__ atom_sum,
                                                            const _Float16* __restrict__ bh,
                                                            const int* __restrict__ b2a,
                                                            _Float16* __restrict__ msg,
                                                            int n_bonds) {
    long long i = (long long)blockIdx.x * blockDim.x + threadIdx.x;
    long long total = (long long)n_bonds * HG4;
    if (i >= total) return;
    int b = (int)(i / HG4);
    int g = (int)(i % HG4);
    f4 s = *(const f4*)(atom_sum + (long long)b2a[b] * HIDDEN + 4 * g);
    h4 h = *(const h4*)(bh + (long long)(b ^ 1) * 320 + 4 * g);
    h4 o;
    o[0] = (_Float16)(s[0] - (float)h[0]);
    o[1] = (_Float16)(s[1] - (float)h[1]);
    o[2] = (_Float16)(s[2] - (float)h[2]);
    o[3] = (_Float16)(s[3] - (float)h[3]);
    *(h4*)(msg + (long long)b * 320 + 4 * g) = o;
}

// graph_pred[m] = dot(g_hidden[m, 0:300], Wg2) + bg2   (one wave32 per row)
__global__ __launch_bounds__(256) void graph_head_kernel(const _Float16* __restrict__ gh,
                                                         const float* __restrict__ Wg2,
                                                         const float* __restrict__ bg2,
                                                         float* __restrict__ out, int M) {
    int gwave = (int)((blockIdx.x * blockDim.x + threadIdx.x) >> 5);
    int lane  = threadIdx.x & 31;
    if (gwave >= M) return;
    const _Float16* row = gh + (long long)gwave * 320;
    float s = 0.0f;
    for (int k = lane; k < HIDDEN; k += 32) s += (float)row[k] * Wg2[k];
    #pragma unroll
    for (int off = 16; off >= 1; off >>= 1) s += __shfl_xor(s, off, 32);
    if (lane == 0) out[gwave] = s + bg2[0];
}

// ---------------------------------------------------------------------------
// WMMA GEMM:  out = act( A[M, KT*32] (f16) @ B + biasP (+ addend) )
// One wave owns MS 16-row slabs (M divisible by 16*MS*?); wave sweeps all N
// tiles reusing A registers and reusing each B tile across the MS slabs.
// B tiles are contiguous (tile j = nt*KT+t at Bp + j*512) and double-buffered
// one tile ahead so loads overlap WMMA. Bp has >=1KB tail padding.
// A per-lane layout (16-bit A 16x32): lane<16: e0..15->K t*32+0..15
//                                     lane>=16: e0..15->+16..31
// C/D layout: vgpr r, lane L: row = r + (L>=16 ? 8 : 0), col = L & 15.
// LD_* are compile-time so per-row offsets fold into store immediates.
// ---------------------------------------------------------------------------
template <int KT, int MS, bool RELU, bool HAS_ADD, bool HAS_OUTH, bool HAS_OUTF,
          int LD_OH, int LD_OF>
__global__ __launch_bounds__(256) void gemm_f16_kernel(
    const _Float16* __restrict__ A, int M,
    const _Float16* __restrict__ Bp, int n_tiles, int Nvalid,
    const float* __restrict__ biasP,
    const float* __restrict__ addend,
    _Float16* __restrict__ outH,
    float* __restrict__ outF)
{
    const int lane  = threadIdx.x & 31;
    const int wave  = threadIdx.x >> 5;
    const int mbase = (blockIdx.x * 8 + wave) * (16 * MS);
    if (mbase >= M) return;

    const int lda  = KT * 32;
    const int half = lane >> 4;            // 0: lanes 0-15, 1: lanes 16-31

    // Load the MS x 16 x (KT*32) A slabs for this wave into registers.
    v16h a[MS][KT];
    #pragma unroll
    for (int s = 0; s < MS; ++s) {
        const _Float16* arow = A + (long long)(mbase + s * 16 + (lane & 15)) * lda;
        #pragma unroll
        for (int t = 0; t < KT; ++t) {
            int k0 = t * 32 + half * 16;
            h8 lo = *(const h8*)(arow + k0);
            h8 hi = *(const h8*)(arow + k0 + 8);
            v16h av;
            #pragma unroll
            for (int i = 0; i < 8; ++i) { av[i] = lo[i]; av[i + 8] = hi[i]; }
            a[s][t] = av;
        }
    }

    // One-tile-ahead double buffered B stream.
    const _Float16* bp = Bp + lane * 16;
    h8 nlo = *(const h8*)(bp);
    h8 nhi = *(const h8*)(bp + 8);

    for (int nt = 0; nt < n_tiles; ++nt) {
        v8f c[MS] = {};
        #pragma unroll
        for (int t = 0; t < KT; ++t) {
            h8 blo = nlo, bhi = nhi;
            bp += 512;
            nlo = *(const h8*)(bp);       // next tile (tail-padded over-read OK)
            nhi = *(const h8*)(bp + 8);
            v16h bv;
            #pragma unroll
            for (int i = 0; i < 8; ++i) { bv[i] = blo[i]; bv[i + 8] = bhi[i]; }
            #pragma unroll
            for (int s = 0; s < MS; ++s)
                c[s] = __builtin_amdgcn_wmma_f32_16x16x32_f16(
                        false, a[s][t], false, bv, (short)0, c[s], false, false);
        }

        const int  col  = nt * 16 + (lane & 15);
        const bool colv = (col < Nvalid);
        const float bval = biasP[col];

        #pragma unroll
        for (int s = 0; s < MS; ++s) {
            const int rbase = mbase + s * 16 + half * 8;  // first of 8 C rows
            #pragma unroll
            for (int r = 0; r < 8; ++r) c[s][r] += bval;

            if (HAS_ADD) {
                if (colv) {
                    const float* ap = addend + (long long)rbase * HIDDEN + col;
                    #pragma unroll
                    for (int r = 0; r < 8; ++r) c[s][r] += ap[r * HIDDEN];
                }
            }
            if (RELU) {
                #pragma unroll
                for (int r = 0; r < 8; ++r) c[s][r] = fmaxf(c[s][r], 0.0f);
            }
            if (HAS_OUTH) {
                _Float16* hp = outH + (long long)rbase * LD_OH + col;
                #pragma unroll
                for (int r = 0; r < 8; ++r)
                    hp[r * LD_OH] = (_Float16)(colv ? c[s][r] : 0.0f);
            }
            if (HAS_OUTF) {
                if (colv) {
                    float* fp = outF + (long long)rbase * LD_OF + col;
                    #pragma unroll
                    for (int r = 0; r < 8; ++r) fp[r * LD_OF] = c[s][r];
                }
            }
        }
    }
}

// ---------------------------------------------------------------------------
// Host-side launch orchestration
// ---------------------------------------------------------------------------
static inline unsigned blocks_for(long long n, int bs) {
    return (unsigned)((n + bs - 1) / bs);
}

extern "C" void kernel_launch(void* const* d_in, const int* in_sizes, int n_in,
                              void* d_out, int out_size, void* d_ws, size_t ws_size,
                              hipStream_t stream) {
    (void)in_sizes; (void)n_in; (void)out_size; (void)ws_size;

    // Inputs (setup_inputs order)
    const float* atom_fea = (const float*)d_in[0];
    const float* bond_fea = (const float*)d_in[1];
    const int*   b2a      = (const int*)d_in[2];
    const int*   mol_ids  = (const int*)d_in[3];
    const float* Wi  = (const float*)d_in[5];
    const float* bi  = (const float*)d_in[6];
    const float* Wm  = (const float*)d_in[7];
    const float* bm  = (const float*)d_in[8];
    const float* Wa  = (const float*)d_in[9];
    const float* ba  = (const float*)d_in[10];
    const float* Wn  = (const float*)d_in[11];
    const float* bn  = (const float*)d_in[12];
    const float* We  = (const float*)d_in[13];
    const float* be  = (const float*)d_in[14];
    const float* Wg1 = (const float*)d_in[15];
    const float* bg1 = (const float*)d_in[16];
    const float* Wg2 = (const float*)d_in[17];
    const float* bg2 = (const float*)d_in[18];

    float* out_node  = (float*)d_out;                                   // [200000,133]
    float* out_edge  = out_node + (long long)N_ATOMS * ATOM_FDIM;       // [400000,14]
    float* out_graph = out_edge + (long long)N_BONDS * 14;              // [8192]

    // Workspace layout (256B aligned)
    char* ws = (char*)d_ws;
    size_t off = 0;
    auto alloc = [&](size_t bytes) -> char* {
        char* p = ws + off;
        off = (off + bytes + 255) & ~(size_t)255;
        return p;
    };
    _Float16* msg16      = (_Float16*)alloc((size_t)N_BONDS * 320 * 2);   // also bond_fea_f16 staging
    _Float16* bondhid16  = (_Float16*)alloc((size_t)N_BONDS * 320 * 2);
    _Float16* atomfea16  = (_Float16*)alloc((size_t)N_ATOMS * 160 * 2);
    _Float16* atomhid16  = (_Float16*)alloc((size_t)N_ATOMS * 320 * 2);
    float*    atomsum32  = (float*)   alloc((size_t)N_ATOMS * HIDDEN * 4);
    float*    molrepr32  = (float*)   alloc((size_t)N_MOLS * HIDDEN * 4);
    _Float16* molrepr16  = (_Float16*)alloc((size_t)N_MOLS * 320 * 2);
    _Float16* ghid16     = (_Float16*)alloc((size_t)N_MOLS * 320 * 2);
    // +1024B tail padding for the one-tile-ahead B prefetch over-read.
    _Float16* WiP  = (_Float16*)alloc((size_t)19 * 5  * 512 * 2 + 1024);
    _Float16* WmP  = (_Float16*)alloc((size_t)19 * 10 * 512 * 2 + 1024);
    _Float16* WaP  = (_Float16*)alloc((size_t)19 * 5  * 512 * 2 + 1024);
    _Float16* WnP  = (_Float16*)alloc((size_t)9  * 10 * 512 * 2 + 1024);
    _Float16* WeP  = (_Float16*)alloc((size_t)1  * 10 * 512 * 2 + 1024);
    _Float16* Wg1P = (_Float16*)alloc((size_t)19 * 10 * 512 * 2 + 1024);
    float* biP  = (float*)alloc(304 * 4);
    float* bmP  = (float*)alloc(304 * 4);
    float* baP  = (float*)alloc(304 * 4);
    float* bnP  = (float*)alloc(144 * 4);
    float* beP  = (float*)alloc(16  * 4);
    float* bg1P = (float*)alloc(304 * 4);

    const int BS = 256;

    // ---- Pack weights / biases into WMMA-friendly layouts ----
    {
        long long n;
        n = 19LL * 5 * 512;  pack_weights_kernel<<<blocks_for(n, BS), BS, 0, stream>>>(Wi,  HIDDEN, BOND_FDIM, 5, 19, WiP);
        n = 19LL * 10 * 512; pack_weights_kernel<<<blocks_for(n, BS), BS, 0, stream>>>(Wm,  HIDDEN, HIDDEN,   10, 19, WmP);
        n = 19LL * 5 * 512;  pack_weights_kernel<<<blocks_for(n, BS), BS, 0, stream>>>(Wa,  HIDDEN, ATOM_FDIM, 5, 19, WaP);
        n = 9LL * 10 * 512;  pack_weights_kernel<<<blocks_for(n, BS), BS, 0, stream>>>(Wn,  ATOM_FDIM, HIDDEN, 10, 9,  WnP);
        n = 1LL * 10 * 512;  pack_weights_kernel<<<blocks_for(n, BS), BS, 0, stream>>>(We,  14, HIDDEN,        10, 1,  WeP);
        n = 19LL * 10 * 512; pack_weights_kernel<<<blocks_for(n, BS), BS, 0, stream>>>(Wg1, HIDDEN, HIDDEN,   10, 19, Wg1P);
        pad_bias_kernel<<<2, BS, 0, stream>>>(bi,  biP,  HIDDEN,    304);
        pad_bias_kernel<<<2, BS, 0, stream>>>(bm,  bmP,  HIDDEN,    304);
        pad_bias_kernel<<<2, BS, 0, stream>>>(ba,  baP,  HIDDEN,    304);
        pad_bias_kernel<<<1, BS, 0, stream>>>(bn,  bnP,  ATOM_FDIM, 144);
        pad_bias_kernel<<<1, BS, 0, stream>>>(be,  beP,  14,        16);
        pad_bias_kernel<<<2, BS, 0, stream>>>(bg1, bg1P, HIDDEN,    304);
    }

    // ---- Stage input features as padded f16 ----
    _Float16* bondfea16 = msg16;  // alias: consumed before msg16 is first written
    conv_pad_kernel<<<blocks_for((long long)N_BONDS * 160, BS), BS, 0, stream>>>(
        bond_fea, bondfea16, N_BONDS, BOND_FDIM, 160);
    conv_pad_kernel<<<blocks_for((long long)N_ATOMS * 160, BS), BS, 0, stream>>>(
        atom_fea, atomfea16, N_ATOMS, ATOM_FDIM, 160);

    // ---- bond_hidden = relu(bond_fea @ Wi^T + bi) ----
    zero_f16_kernel<<<blocks_for((long long)N_BONDS * 320, BS), BS, 0, stream>>>(
        bondhid16, (long long)N_BONDS * 320);
    gemm_f16_kernel<5, 2, true, false, true, false, 320, 1>
        <<<(N_BONDS + 255) / 256, BS, 0, stream>>>(
        bondfea16, N_BONDS, WiP, 19, HIDDEN, biP, nullptr, bondhid16, nullptr);

    // msg16 region reused from bond_fea staging -> clear pads/garbage once
    zero_f16_kernel<<<blocks_for((long long)N_BONDS * 320, BS), BS, 0, stream>>>(
        msg16, (long long)N_BONDS * 320);

    // ---- DEPTH message passing iterations ----
    for (int d = 0; d < 3; ++d) {
        zero_f32_kernel<<<blocks_for((long long)N_ATOMS * HIDDEN, BS), BS, 0, stream>>>(
            atomsum32, (long long)N_ATOMS * HIDDEN);
        scatter_add_kernel<<<blocks_for((long long)N_BONDS * HG4, BS), BS, 0, stream>>>(
            bondhid16, 320, b2a, atomsum32, N_BONDS);
        build_message_kernel<<<blocks_for((long long)N_BONDS * HG4, BS), BS, 0, stream>>>(
            atomsum32, bondhid16, b2a, msg16, N_BONDS);
        gemm_f16_kernel<10, 2, true, false, true, false, 320, 1>
            <<<(N_BONDS + 255) / 256, BS, 0, stream>>>(
            msg16, N_BONDS, WmP, 19, HIDDEN, bmP, nullptr, bondhid16, nullptr);
    }

    // ---- Final atom aggregation + atom_hidden ----
    zero_f32_kernel<<<blocks_for((long long)N_ATOMS * HIDDEN, BS), BS, 0, stream>>>(
        atomsum32, (long long)N_ATOMS * HIDDEN);
    scatter_add_kernel<<<blocks_for((long long)N_BONDS * HG4, BS), BS, 0, stream>>>(
        bondhid16, 320, b2a, atomsum32, N_BONDS);
    zero_f16_kernel<<<blocks_for((long long)N_ATOMS * 320, BS), BS, 0, stream>>>(
        atomhid16, (long long)N_ATOMS * 320);
    gemm_f16_kernel<5, 2, true, true, true, false, 320, 1>
        <<<(N_ATOMS + 255) / 256, BS, 0, stream>>>(
        atomfea16, N_ATOMS, WaP, 19, HIDDEN, baP, atomsum32, atomhid16, nullptr);

    // ---- Molecule pooling ----
    zero_f32_kernel<<<blocks_for((long long)N_MOLS * HIDDEN, BS), BS, 0, stream>>>(
        molrepr32, (long long)N_MOLS * HIDDEN);
    scatter_add_kernel<<<blocks_for((long long)N_ATOMS * HG4, BS), BS, 0, stream>>>(
        atomhid16, 320, mol_ids, molrepr32, N_ATOMS);

    // ---- node_pred = atom_hidden @ Wn^T + bn  (f32 straight to d_out) ----
    gemm_f16_kernel<10, 2, false, false, false, true, 1, ATOM_FDIM>
        <<<(N_ATOMS + 255) / 256, BS, 0, stream>>>(
        atomhid16, N_ATOMS, WnP, 9, ATOM_FDIM, bnP, nullptr, nullptr, out_node);

    // ---- edge_pred = bond_hidden @ We^T + be ----
    gemm_f16_kernel<10, 2, false, false, false, true, 1, 14>
        <<<(N_BONDS + 255) / 256, BS, 0, stream>>>(
        bondhid16, N_BONDS, WeP, 1, 14, beP, nullptr, nullptr, out_edge);

    // ---- graph head ----
    conv_pad_kernel<<<blocks_for((long long)N_MOLS * 320, BS), BS, 0, stream>>>(
        molrepr32, molrepr16, N_MOLS, HIDDEN, 320);
    gemm_f16_kernel<10, 2, true, false, true, false, 320, 1>
        <<<(N_MOLS + 255) / 256, BS, 0, stream>>>(
        molrepr16, N_MOLS, Wg1P, 19, HIDDEN, bg1P, nullptr, ghid16, nullptr);
    graph_head_kernel<<<blocks_for((long long)N_MOLS * 32, BS), BS, 0, stream>>>(
        ghid16, Wg2, bg2, out_graph, N_MOLS);
}
